// model_38165079392362
// MI455X (gfx1250) — compile-verified, matrix operand order checked by reading, not run on hardware
//
#include <hip/hip_runtime.h>
#include <hip/hip_bf16.h>
#include <math.h>

typedef __bf16 bf16_t;
typedef __attribute__((ext_vector_type(8)))  bf16_t v8bf;
typedef __attribute__((ext_vector_type(16))) bf16_t v16bf;
typedef __attribute__((ext_vector_type(8)))  float  v8f;
typedef __attribute__((ext_vector_type(4)))  int    i32x4;

#define SEQ   512
#define BATCH 64
#define INF_  1024   // IN
#define HID   1024
#define NWG   32     // persistent workgroups for the scan

// ---------- gfx1250 async global->LDS copy (ASYNCcnt path), with fallback ----------
#if defined(__AMDGCN__) && __has_builtin(__builtin_amdgcn_global_load_async_to_lds_b128)
#define HAVE_ASYNC_LDS 1
#else
#define HAVE_ASYNC_LDS 0
#endif

__device__ __forceinline__ void copy16_g2l(const bf16_t* g, bf16_t* l) {
#if HAVE_ASYNC_LDS
  __builtin_amdgcn_global_load_async_to_lds_b128(
      (__attribute__((address_space(1))) i32x4*)g,
      (__attribute__((address_space(3))) i32x4*)l, 0, 0);
#else
  *(v8bf*)l = *(const v8bf*)g;
#endif
}

__device__ __forceinline__ void async_wait_all() {
#if HAVE_ASYNC_LDS
#if __has_builtin(__builtin_amdgcn_s_wait_asynccnt)
  __builtin_amdgcn_s_wait_asynccnt(0);
#else
  asm volatile("s_wait_asynccnt 0x0" ::: "memory");
#endif
#endif
}

// ---------- fragment helpers ----------
__device__ __forceinline__ v16bf make_frag(v8bf a, v8bf b) {
  return __builtin_shufflevector(a, b, 0,1,2,3,4,5,6,7,8,9,10,11,12,13,14,15);
}

// A/B fragment from a row-major (rows x ld) bf16 matrix.
// ISA 16-bit 16x32 layout: lane%16 = row; lanes>=16 shift K by 8; halves at K and K+16.
__device__ __forceinline__ v16bf load_frag_g(const bf16_t* __restrict__ base,
                                             int ld, int row0, int k0) {
  int lane = threadIdx.x & 31;
  int r    = lane & 15;
  int klo  = (lane >> 4) << 3;
  const bf16_t* p = base + (size_t)(row0 + r) * ld + (k0 + klo);
  return make_frag(*(const v8bf*)p, *(const v8bf*)(p + 16));
}

// ---------- init kernels ----------
__global__ void f32_to_bf16(const float* __restrict__ src, bf16_t* __restrict__ dst,
                            size_t n) {
  size_t i = (size_t)blockIdx.x * blockDim.x + threadIdx.x;
  size_t stride = (size_t)gridDim.x * blockDim.x;
  for (; i < n; i += stride) dst[i] = (bf16_t)src[i];
}

__global__ void zero_int(int* p) { if (threadIdx.x == 0) *p = 0; }

// ---------- Phase 1: x_proj = X @ W_ih^T + (b_ih + b_hh) ----------
// block = 256 threads (8 waves). WG tile: 128 rows x 64 cols.
// W_ih tile (64 x 32 bf16 = 4 KB) double-buffered in LDS via async copies;
// each wave: 16 rows x 64 cols = 4 accumulator frags, A frag reused 4x.
__global__ __launch_bounds__(256)
void xproj_gemm(const bf16_t* __restrict__ Xbf,   // (32768, 1024) row-major
                const bf16_t* __restrict__ Wih,   // (1024, 1024) row-major (N,K)
                const float*  __restrict__ b_ih,
                const float*  __restrict__ b_hh,
                float*        __restrict__ out) { // (32768, 1024)
  __shared__ bf16_t Bs[2][64 * 32];               // 8 KB double buffer

  const int wave = threadIdx.x >> 5;
  const int lane = threadIdx.x & 31;
  const int m0 = blockIdx.x * 128 + wave * 16;
  const int n0 = blockIdx.y * 64;
  const int r    = lane & 15;
  const int klo  = (lane >> 4) << 3;

  // staging assignment: 256 threads x 16B = exactly one 64x32 tile
  const int srow = threadIdx.x >> 2;        // 0..63
  const int sseg = (threadIdx.x & 3) * 8;   // 0,8,16,24
  const bf16_t* wbase = Wih + (size_t)(n0 + srow) * INF_ + sseg;

  copy16_g2l(wbase + 0, &Bs[0][srow * 32 + sseg]);

  v8f acc[4] = {};
  for (int k0 = 0; k0 < INF_; k0 += 32) {
    const int cur = (k0 >> 5) & 1;
    async_wait_all();
    __syncthreads();                               // Bs[cur] ready for all waves
    if (k0 + 32 < INF_)
      copy16_g2l(wbase + k0 + 32, &Bs[cur ^ 1][srow * 32 + sseg]);

    v16bf a = load_frag_g(Xbf, INF_, m0, k0);
#pragma unroll
    for (int nj = 0; nj < 4; ++nj) {
      const bf16_t* pb = &Bs[cur][(nj * 16 + r) * 32 + klo];
      v16bf b = make_frag(*(const v8bf*)pb, *(const v8bf*)(pb + 16));
      acc[nj] = __builtin_amdgcn_wmma_f32_16x16x32_bf16(
          false, a, false, b, (short)0, acc[nj], false, false);
    }
  }

  const int rbase = (lane >> 4) * 8;
#pragma unroll
  for (int nj = 0; nj < 4; ++nj) {
    int n = n0 + nj * 16 + r;
    float bias = b_ih[n] + b_hh[n];
#pragma unroll
    for (int rr = 0; rr < 8; ++rr) {
      int m = m0 + rbase + rr;
      out[(size_t)m * HID + n] = acc[nj][rr] + bias;
    }
  }
}

// ---------- Phase 2: persistent recurrent scan ----------
// 32 WGs x 256 threads. WG g owns output columns [g*32, g*32+32).
// W_hh slice (32 rows x 1024 K, bf16 = 64 KB) cached in LDS (async staged).
// 8 waves = 4 M-tiles (rows of h, 64 = 4x16) x 2 N-subtiles; 2 accumulators
// interleaved over K to break the WMMA RAW chain.
__global__ __launch_bounds__(256)
void rnn_scan(const bf16_t* __restrict__ Whh,   // (1024,1024) row-major (N,K)
              bf16_t*       __restrict__ hbuf,  // 2 * BATCH*HID ping-pong
              float*        __restrict__ out,   // outs then h_last tail
              int*          __restrict__ cnt) {
  __shared__ bf16_t Wsl[32 * HID];              // 64 KB

  const int wg = blockIdx.x;
  {
    const bf16_t* wsrc = Whh + (size_t)wg * 32 * HID;
    for (int i = threadIdx.x * 8; i < 32 * HID; i += 256 * 8)
      copy16_g2l(&wsrc[i], &Wsl[i]);
    async_wait_all();
  }
  __syncthreads();

  const int wave = threadIdx.x >> 5;
  const int lane = threadIdx.x & 31;
  const int mi   = wave & 3;        // row tile: rows mi*16 .. mi*16+15
  const int ni   = wave >> 2;       // column subtile within slice (0/1)
  const int r    = lane & 15;
  const int klo  = (lane >> 4) << 3;
  const int rbase = (lane >> 4) * 8;
  const int n    = wg * 32 + ni * 16 + r;       // global output column

  for (int t = 0; t < SEQ; ++t) {
    const bf16_t* hprev = hbuf + (size_t)(t & 1) * (BATCH * HID);
    bf16_t*       hnext = hbuf + (size_t)((t + 1) & 1) * (BATCH * HID);

    v8f acc0 = {}, acc1 = {};
#pragma unroll 2
    for (int k0 = 0; k0 < HID; k0 += 64) {
      const bf16_t* pa = hprev + (size_t)(mi * 16 + r) * HID + k0 + klo;
      const bf16_t* pb = &Wsl[(size_t)(ni * 16 + r) * HID + k0 + klo];
      v16bf a0 = make_frag(*(const v8bf*)pa, *(const v8bf*)(pa + 16));
      v16bf b0 = make_frag(*(const v8bf*)pb, *(const v8bf*)(pb + 16));
      v16bf a1 = make_frag(*(const v8bf*)(pa + 32), *(const v8bf*)(pa + 48));
      v16bf b1 = make_frag(*(const v8bf*)(pb + 32), *(const v8bf*)(pb + 48));
      acc0 = __builtin_amdgcn_wmma_f32_16x16x32_bf16(
          false, a0, false, b0, (short)0, acc0, false, false);
      acc1 = __builtin_amdgcn_wmma_f32_16x16x32_bf16(
          false, a1, false, b1, (short)0, acc1, false, false);
    }
    v8f acc = acc0 + acc1;

    float* orow = out + (size_t)t * (BATCH * HID);
#pragma unroll
    for (int rr = 0; rr < 8; ++rr) {
      int m = mi * 16 + rbase + rr;
      size_t idx = (size_t)m * HID + n;
      float v = tanhf(acc[rr] + orow[idx]);     // x_proj read, h_t written back
      orow[idx]  = v;
      hnext[idx] = (bf16_t)v;
      if (t == SEQ - 1)
        out[(size_t)SEQ * BATCH * HID + idx] = v;   // h_n tail
    }

    // device-wide barrier (monotonic count; cnt zeroed per call -> replay safe)
    __threadfence();
    __syncthreads();
    if (threadIdx.x == 0) {
      atomicAdd(cnt, 1);
      while (__hip_atomic_load(cnt, __ATOMIC_ACQUIRE, __HIP_MEMORY_SCOPE_AGENT) <
             NWG * (t + 1)) {
        __builtin_amdgcn_s_sleep(1);
      }
    }
    __syncthreads();
    __threadfence();
  }
}

// ---------- launch ----------
extern "C" void kernel_launch(void* const* d_in, const int* in_sizes, int n_in,
                              void* d_out, int out_size, void* d_ws, size_t ws_size,
                              hipStream_t stream) {
  (void)in_sizes; (void)n_in; (void)out_size; (void)ws_size;
  const float* input  = (const float*)d_in[0];  // (512,64,1024)
  const float* hidden = (const float*)d_in[1];  // (1,64,1024)
  const float* W_ih   = (const float*)d_in[2];  // (1024,1024)
  const float* W_hh   = (const float*)d_in[3];  // (1024,1024)
  const float* b_ih   = (const float*)d_in[4];
  const float* b_hh   = (const float*)d_in[5];
  float* out = (float*)d_out;

  // workspace layout
  char* ws = (char*)d_ws;
  size_t off = 0;
  bf16_t* Xbf  = (bf16_t*)(ws + off); off += (size_t)SEQ * BATCH * INF_ * sizeof(bf16_t);
  bf16_t* Wihb = (bf16_t*)(ws + off); off += (size_t)HID * INF_ * sizeof(bf16_t);
  bf16_t* Whhb = (bf16_t*)(ws + off); off += (size_t)HID * HID * sizeof(bf16_t);
  bf16_t* hbuf = (bf16_t*)(ws + off); off += (size_t)2 * BATCH * HID * sizeof(bf16_t);
  off = (off + 255) & ~(size_t)255;
  int* cnt = (int*)(ws + off);

  // convert to bf16
  f32_to_bf16<<<4096, 256, 0, stream>>>(input,  Xbf,  (size_t)SEQ * BATCH * INF_);
  f32_to_bf16<<<1024, 256, 0, stream>>>(W_ih,   Wihb, (size_t)HID * INF_);
  f32_to_bf16<<<1024, 256, 0, stream>>>(W_hh,   Whhb, (size_t)HID * HID);
  f32_to_bf16<<<64,   256, 0, stream>>>(hidden, hbuf, (size_t)BATCH * HID);  // h0 -> buf 0
  zero_int<<<1, 64, 0, stream>>>(cnt);

  // Phase 1: x_proj (+biases) into d_out outs region
  dim3 g1(SEQ * BATCH / 128, HID / 64);
  xproj_gemm<<<g1, 256, 0, stream>>>(Xbf, Wihb, b_ih, b_hh, out);

  // Phase 2: persistent scan, rewrites outs in place, writes h_n tail
  rnn_scan<<<NWG, 256, 0, stream>>>(Whhb, hbuf, out, cnt);
}
